// NestedAttention_8950711845473
// MI455X (gfx1250) — matrix operand order repaired and checked by reading, not compiled, not run on hardware
//
#include <hip/hip_runtime.h>

typedef __attribute__((ext_vector_type(16))) _Float16 v16h;
typedef __attribute__((ext_vector_type(8)))  _Float16 v8h;
typedef __attribute__((ext_vector_type(2)))  _Float16 v2h;
typedef __attribute__((ext_vector_type(8)))  float    v8f;
typedef __attribute__((ext_vector_type(4)))  int      v4i;

#define B_  4
#define C_  256
#define N_  2304
#define R_  64
#define NW  8            // waves per block in attention kernel
#define MTILES (N_ / 32) // 72 m-tiles per j
#define NSTEP  (3 * MTILES)
// q is pre-scaled by 1/sqrt(64) * log2(e) so softmax runs in exp2 domain
#define QSCALE_ (0.125f * 1.4426950408889634f)
#define LOG2E_  1.4426950408889634f

#if __has_builtin(__builtin_amdgcn_global_load_async_to_lds_b128)
#define HAVE_ASYNC_LDS 1
#else
#define HAVE_ASYNC_LDS 0
#endif

// raw v_exp_f32 (no denormal-range fixup; fine for softmax weights)
static __device__ __forceinline__ float fast_exp2(float x) {
#if __has_builtin(__builtin_amdgcn_exp2f)
    return __builtin_amdgcn_exp2f(x);
#else
    return exp2f(x);
#endif
}

// builtin signature (from compiler diagnostic): AS1 int4* src, AS3 int4* dst
typedef __attribute__((address_space(1))) v4i* glb_v4i_p;
typedef __attribute__((address_space(3))) v4i* lds_v4i_p;

static __device__ __forceinline__ void copy16_to_lds(void* ldst, const void* gsrc) {
#if HAVE_ASYNC_LDS
    // async DMA: 16B per lane, tracked by ASYNCcnt
    __builtin_amdgcn_global_load_async_to_lds_b128((glb_v4i_p)gsrc, (lds_v4i_p)ldst, 0, 0);
#else
    *(v8h*)ldst = *(const v8h*)gsrc;
#endif
}
static __device__ __forceinline__ void wait_async_all() {
#if HAVE_ASYNC_LDS
 #if __has_builtin(__builtin_amdgcn_s_wait_asynccnt)
    __builtin_amdgcn_s_wait_asynccnt(0);
 #else
    asm volatile("s_wait_asynccnt 0" ::: "memory");
 #endif
#endif
}

static __device__ __forceinline__ v16h cat8(v8h a, v8h b) {
    return __builtin_shufflevector(a, b, 0,1,2,3,4,5,6,7,8,9,10,11,12,13,14,15);
}
static __device__ __forceinline__ v8f zero8() {
    v8f z;
#pragma unroll
    for (int e = 0; e < 8; ++e) z[e] = 0.f;
    return z;
}
static __device__ __forceinline__ v8f wmma_f16(v16h a, v16h b, v8f c) {
    // D = A(16x32 f16) * B(32x16 f16) + C(16x16 f32)
    return __builtin_amdgcn_wmma_f32_16x16x32_f16(false, a, false, b, (short)0, c, false, false);
}

// ---------------------------------------------------------------------------
// Stage 1: q/k/v projections (9 small GEMMs), fp32 math, f16 outputs.
// q,k stored [i][b][n][r] (row of r contiguous); q pre-scaled by QSCALE_.
// v stored [i][b][r][n] with columns permuted within each 32-block:
//   logical u = n%32 -> physical 2*(u%16) + (u/16)
// so the attention kernel's packed-P K ordering matches V's K ordering.
// ---------------------------------------------------------------------------
__global__ void na_proj(const float* __restrict__ x,
                        const float* __restrict__ wq,
                        const float* __restrict__ wk,
                        const float* __restrict__ wv,
                        _Float16* __restrict__ qh,
                        _Float16* __restrict__ kh,
                        _Float16* __restrict__ vh) {
    const int n   = blockIdx.x * blockDim.x + threadIdx.x;  // 0..N_-1
    const int rr  = blockIdx.y;                             // 0..63
    const int z   = blockIdx.z;                             // 0..35
    const int mat = z / 12;
    const int i   = (z / B_) % 3;
    const int b   = z % B_;

    const float* w  = (mat == 0 ? wq : (mat == 1 ? wk : wv)) + (size_t)(i * R_ + rr) * C_;
    const float* xb = x + (size_t)b * C_ * N_ + n;

    float acc = 0.f;
#pragma unroll 8
    for (int c = 0; c < C_; ++c)
        acc = fmaf(w[c], xb[(size_t)c * N_], acc);

    if (mat == 0) {
        qh[(((size_t)(i * B_ + b) * N_) + n) * R_ + rr] = (_Float16)(acc * QSCALE_);
    } else if (mat == 1) {
        kh[(((size_t)(i * B_ + b) * N_) + n) * R_ + rr] = (_Float16)acc;
    } else {
        const int u    = n & 31;
        const int npos = (n & ~31) | ((u & 15) << 1) | (u >> 4);
        vh[(((size_t)(i * B_ + b) * R_) + rr) * N_ + npos] = (_Float16)acc;
    }
}

// ---------------------------------------------------------------------------
// Stage 2: fused nested attention, flash-style, lazy-max online softmax.
// Block = 8 waves, all on the same (b,i), each owning a 16-row n-tile.
// K/V tiles are staged once per block into double-buffered LDS with async
// global->LDS loads (ASYNCcnt) and consumed by all 8 waves -> 8x less
// global/L2 traffic. Nested loops (j outer, m-tiles inner) so flash-state
// reset/finalize stay out of the hot loop (no if-conversion), with the
// prefetch index flattened across j boundaries.
// grid: x = N/(16*NW), y = 3 (i), z = B; block = 32*NW.
// ---------------------------------------------------------------------------
__global__ void __launch_bounds__(32 * NW, 1)
na_attn(const _Float16* __restrict__ qh,
        const _Float16* __restrict__ kh,
        const _Float16* __restrict__ vh,
        float* __restrict__ comb) {
    __shared__ __align__(16) _Float16 kbuf[2][32][R_];   // 2 x 4KB  [m-row][r]
    __shared__ __align__(16) _Float16 vbuf[2][R_][32];   // 2 x 4KB  [r][m-col']
    __shared__ __align__(16) _Float16 pbuf[NW][16][32];  // 8KB per-wave P staging

    const int tid  = threadIdx.x;
    const int lane = tid & 31;
    const int wave = tid >> 5;
    const int ln   = lane & 15;   // column lane within 16
    const int grp  = lane >> 4;   // lane half-group
    const int n0   = (blockIdx.x * NW + wave) * 16;
    const int i    = blockIdx.y;
    const int b    = blockIdx.z;

    // cooperative K/V tile staging for flat step s into buffer buf
    auto stage = [&](int buf, int s) {
        const int sj = s / MTILES;
        const int m0 = (s % MTILES) * 32;
        const _Float16* kg = kh + ((size_t)(sj * B_ + b) * N_ + m0) * R_;  // contiguous 4KB
        const _Float16* vg = vh + ((size_t)(sj * B_ + b) * R_) * N_ + m0;  // 64 rows x 64B
        copy16_to_lds(&kbuf[buf][0][0] + tid * 8, kg + tid * 8);
        const int r  = tid >> 2;
        const int ch = (tid & 3) * 8;
        copy16_to_lds(&vbuf[buf][r][ch], vg + (size_t)r * N_ + ch);
    };

    // Q^T tile (16 n-rows x 64 r) in WMMA A-layout, two K=32 chunks.
    // A order used here (B matches): lane row M = lane%16, elem e -> K = grp*16 + e.
    const _Float16* qptr = qh + (((size_t)(i * B_ + b) * N_) + n0 + ln) * R_;
    v16h qa[2];
#pragma unroll
    for (int kk = 0; kk < 2; ++kk) {
        v8h lo = *(const v8h*)(qptr + kk * 32 + grp * 16);
        v8h hi = *(const v8h*)(qptr + kk * 32 + grp * 16 + 8);
        qa[kk] = cat8(lo, hi);
    }

    v8f Ofin[4];
#pragma unroll
    for (int rc = 0; rc < 4; ++rc) Ofin[rc] = zero8();

    stage(0, 0);
    wait_async_all();
    __syncthreads();

    int s   = 0;
    int cur = 0;
    for (int j = 0; j < 3; ++j) {
        // flash state reset: real code outside the hot loop
        v8f O[4];
        float rm[8], rs[8];
#pragma unroll
        for (int rc = 0; rc < 4; ++rc) O[rc] = zero8();
#pragma unroll
        for (int c = 0; c < 8; ++c) { rm[c] = -1.0e30f; rs[c] = 0.f; }

        for (int mt = 0; mt < MTILES; ++mt, ++s) {
            // prefetch next tile pair (wrap last step to 0: harmless, branchless)
            int nxt = s + 1;
            if (nxt == NSTEP) nxt = 0;
            stage(cur ^ 1, nxt);

            // ---- S = (Q^T K) for a 16 x 32 block (exp2 domain) ------------
            v8f S[2];
#pragma unroll
            for (int t = 0; t < 2; ++t) {
                S[t] = zero8();
#pragma unroll
                for (int kk = 0; kk < 2; ++kk) {
                    v16h kb = *(const v16h*)(&kbuf[cur][t * 16 + ln][kk * 32 + grp * 16]);
                    S[t] = wmma_f16(qa[kk], kb, S[t]);
                }
            }

            // ---- lazy row-max update (rare, wave-uniform slow path) -------
            float mx[8];
            int need = 0;
#pragma unroll
            for (int c = 0; c < 8; ++c) {
                mx[c] = fmaxf(S[0][c], S[1][c]);
                need |= (mx[c] > rm[c]) ? 1 : 0;
            }
            if (__any(need)) {
#pragma unroll
                for (int c = 0; c < 8; ++c) {
                    float m2 = fmaxf(mx[c], rm[c]);
#pragma unroll
                    for (int off = 1; off < 16; off <<= 1)
                        m2 = fmaxf(m2, __shfl_xor(m2, off, 16));
                    const float corr = fast_exp2(rm[c] - m2);
                    rm[c] = m2;
                    rs[c] *= corr;
#pragma unroll
                    for (int rc = 0; rc < 4; ++rc) O[rc][c] *= corr;
                }
            }

            // ---- P = exp2(S - rm), per-lane partial sums, packed staging --
#pragma unroll
            for (int c = 0; c < 8; ++c) {
                const float p0 = fast_exp2(S[0][c] - rm[c]);
                const float p1 = fast_exp2(S[1][c] - rm[c]);
                rs[c] += p0 + p1;
                v2h pk;
                pk[0] = (_Float16)p0;
                pk[1] = (_Float16)p1;
                // packed K slots 2*ln, 2*ln+1 <-> logical cols ln, 16+ln
                *(v2h*)(&pbuf[wave][c + 8 * grp][2 * ln]) = pk;
            }

            // same-wave cross-lane LDS dependency
            asm volatile("s_wait_dscnt 0" ::: "memory");

            v8h plo = *(const v8h*)(&pbuf[wave][ln][grp * 16]);
            v8h phi = *(const v8h*)(&pbuf[wave][ln][grp * 16 + 8]);
            const v16h pa = cat8(plo, phi);

            // ---- O += P (16x32) * V^T (32x64); V stored K-permuted --------
#pragma unroll
            for (int rc = 0; rc < 4; ++rc) {
                v16h vb = *(const v16h*)(&vbuf[cur][rc * 16 + ln][grp * 16]);
                O[rc] = wmma_f16(pa, vb, O[rc]);
            }

            wait_async_all();     // prefetched tiles landed
            __syncthreads();      // all waves done reading cur
            cur ^= 1;
        }

        // ---- j done: one cross-lane sum reduction per row, fold -----------
#pragma unroll
        for (int c = 0; c < 8; ++c) {
            float t = rs[c];
#pragma unroll
            for (int off = 1; off < 16; off <<= 1)
                t += __shfl_xor(t, off, 16);
            const float inv = 1.f / t;
#pragma unroll
            for (int rc = 0; rc < 4; ++rc) Ofin[rc][c] += O[rc][c] * inv;
        }
    }

    // write comb[b][i*64 + r][n]; tile element = row n0+c+8*grp, col rc*16+ln
#pragma unroll
    for (int rc = 0; rc < 4; ++rc) {
#pragma unroll
        for (int c = 0; c < 8; ++c) {
            comb[((size_t)(b * 192 + i * 64 + rc * 16 + ln)) * N_
                 + n0 + c + 8 * grp] = Ofin[rc][c];
        }
    }
}

// ---------------------------------------------------------------------------
// Stage 3: out = x * sigmoid(wo @ comb)
// ---------------------------------------------------------------------------
__global__ void na_out(const float* __restrict__ x,
                       const float* __restrict__ wo,
                       const float* __restrict__ comb,
                       float* __restrict__ out) {
    const int n = blockIdx.x * blockDim.x + threadIdx.x;
    const int c = blockIdx.y;
    const int b = blockIdx.z;

    const float* wrow = wo + (size_t)c * 192;
    const float* cb   = comb + (size_t)b * 192 * N_ + n;

    float acc = 0.f;
#pragma unroll 8
    for (int q = 0; q < 192; ++q)
        acc = fmaf(wrow[q], cb[(size_t)q * N_], acc);

    // sigmoid via raw v_exp_f32: 1/(1+2^(-x*log2e))
    const float e = fast_exp2(-acc * LOG2E_);
    const size_t idx = ((size_t)(b * C_ + c)) * N_ + n;
    out[idx] = x[idx] * (1.f / (1.f + e));
}

// ---------------------------------------------------------------------------
extern "C" void kernel_launch(void* const* d_in, const int* in_sizes, int n_in,
                              void* d_out, int out_size, void* d_ws, size_t ws_size,
                              hipStream_t stream) {
    const float* x  = (const float*)d_in[0];
    const float* wq = (const float*)d_in[1];
    const float* wk = (const float*)d_in[2];
    const float* wv = (const float*)d_in[3];
    const float* wo = (const float*)d_in[4];
    float* out = (float*)d_out;

    const size_t QKV = (size_t)3 * B_ * N_ * R_;   // elements per q/k/v block
    _Float16* qh = (_Float16*)d_ws;
    _Float16* kh = qh + QKV;
    _Float16* vh = kh + QKV;
    float* comb  = (float*)(vh + QKV);             // [B][192][N] f32

    na_proj<<<dim3(N_ / 256, R_, 36), 256, 0, stream>>>(x, wq, wk, wv, qh, kh, vh);
    na_attn<<<dim3(N_ / (16 * NW), 3, B_), 32 * NW, 0, stream>>>(qh, kh, vh, comb);
    na_out <<<dim3(N_ / 256, C_, B_), 256, 0, stream>>>(x, wo, comb, out);
}
